// LSTM_68719477039
// MI455X (gfx1250) — matrix operand order repaired and checked by reading, not compiled, not run on hardware
//
#include <hip/hip_runtime.h>
#include <hip/hip_bf16.h>

// ---------------- problem constants ----------------
#define SEQ   1024
#define BATCH 64
#define IN_   512
#define H_    512
#define KTOT  (IN_ + H_)      // 1024
#define G4H   (4 * H_)        // 2048
#define NWG   32              // persistent workgroups; wg k owns h-cols [16k,16k+16)
#define THREADS 256           // 8 wave32 waves

typedef __attribute__((ext_vector_type(16))) __bf16 v16bf;
typedef __attribute__((ext_vector_type(8)))  float  v8f;

union ABfrag { uint4 u[2]; v16bf v; };   // 32 bytes = 16 bf16 per lane

__device__ __forceinline__ unsigned short f2bf(float f) {
    unsigned u = __float_as_uint(f);
    unsigned r = (u + 0x7FFFu + ((u >> 16) & 1u)) >> 16;   // round-to-nearest-even
    return (unsigned short)r;
}
__device__ __forceinline__ unsigned pack2(float a, float b) {
    return (unsigned)f2bf(a) | ((unsigned)f2bf(b) << 16);
}
__device__ __forceinline__ float sigm(float x) {
    return 1.0f / (1.0f + __expf(-x));
}
__device__ __forceinline__ float tanh_fast(float x) {
    float e = __expf(-2.0f * x);
    return 2.0f / (1.0f + e) - 1.0f;     // saturates correctly at +/-1
}

// ---------------- phase 1: x -> bf16, init barrier ----------------
__global__ void lstm_convert_x(const float* __restrict__ x,
                               unsigned short* __restrict__ xb,
                               unsigned* __restrict__ bar, int n8) {
    int i = blockIdx.x * blockDim.x + threadIdx.x;
    if (i == 0) { bar[0] = 0u; bar[1] = 0u; }
    if (i < n8) {
        const float4* src = (const float4*)x;
        float4 a = src[2 * i];
        float4 b = src[2 * i + 1];
        uint4 o;
        o.x = pack2(a.x, a.y); o.y = pack2(a.z, a.w);
        o.z = pack2(b.x, b.y); o.w = pack2(b.z, b.w);
        ((uint4*)xb)[i] = o;
    }
}

// ---------------- global (grid-wide) barrier ----------------
__device__ __forceinline__ void grid_barrier(unsigned* cnt, unsigned* gen) {
    __syncthreads();
    if (threadIdx.x == 0) {
        __builtin_amdgcn_fence(__ATOMIC_RELEASE, "agent");
        unsigned g = __hip_atomic_load(gen, __ATOMIC_RELAXED, __HIP_MEMORY_SCOPE_AGENT);
        if (__hip_atomic_fetch_add(cnt, 1u, __ATOMIC_ACQ_REL, __HIP_MEMORY_SCOPE_AGENT)
            == NWG - 1u) {
            __hip_atomic_store(cnt, 0u, __ATOMIC_RELAXED, __HIP_MEMORY_SCOPE_AGENT);
            __hip_atomic_fetch_add(gen, 1u, __ATOMIC_RELEASE, __HIP_MEMORY_SCOPE_AGENT);
        } else {
            while (__hip_atomic_load(gen, __ATOMIC_ACQUIRE, __HIP_MEMORY_SCOPE_AGENT) == g) {
                __builtin_amdgcn_s_sleep(2);
            }
        }
        __builtin_amdgcn_fence(__ATOMIC_ACQUIRE, "agent");
    }
    __syncthreads();
}

// ---------------- phase 2: persistent sequential LSTM ----------------
// LDS layout (dynamic):
//   sW : 128 KB  B-fragments of W slice, 128 frags (4 gates x 32 ksteps) x 1 KB
//   sG : 16 KB   gate tiles f32 [64 rows][64 cols]
//   sC : 4 KB    cell state slice f32 [64][16]
//   sB : 256 B   bias slice f32 [64]
__global__ void lstm_seq_wmma(const float* __restrict__ W,
                              const float* __restrict__ bias,
                              const float* __restrict__ h0,
                              const float* __restrict__ c0,
                              const unsigned short* __restrict__ Xb,   // [SEQ][B][IN] bf16
                              unsigned short* __restrict__ hbuf,       // [2][B][H]  bf16
                              float* __restrict__ out,                 // outputs|hT|cT
                              unsigned* __restrict__ bar) {
    extern __shared__ char smem[];
    unsigned short* sW = (unsigned short*)smem;                 // 131072 B
    float* sG = (float*)(smem + 131072);                        // 16384 B
    float* sC = (float*)(smem + 131072 + 16384);                // 4096 B
    float* sB = (float*)(smem + 131072 + 16384 + 4096);         // 256 B

    const int wg   = blockIdx.x;       // 0..31
    const int tid  = threadIdx.x;
    const int lane = tid & 31;
    const int wave = tid >> 5;         // 0..7

    // ---- one-time init: pack W slice into LDS B-fragment layout ----
    // global gate column for (g, c): g*512 + wg*16 + c ; local col n = g*16 + c
    // fragment id = g*32 + kstep ; element (K, n):
    //   lane = ((K&8) == 0) ? (n&15) : 16 + (n&15) ; elem = (K & 7) + ((K >> 4) & 1) * 8
    for (int idx = tid; idx < KTOT * 64; idx += THREADS) {
        int n = idx & 63;
        int K = idx >> 6;
        int g = n >> 4, c = n & 15;
        float w = W[(size_t)K * G4H + g * H_ + wg * 16 + c];
        int frag = g * 32 + (K >> 5);
        int l = ((K & 8) == 0) ? c : (16 + c);
        sW[frag * 512 + l * 16 + (K & 7) + ((K >> 4) & 1) * 8] = f2bf(w);
    }
    for (int idx = tid; idx < 64; idx += THREADS) {
        int g = idx >> 4, c = idx & 15;
        sB[idx] = bias[g * H_ + wg * 16 + c];
    }
    for (int idx = tid; idx < BATCH * 16; idx += THREADS) {
        int r = idx >> 4, c = idx & 15;
        sC[idx] = c0[r * H_ + wg * 16 + c];
        hbuf[r * H_ + wg * 16 + c] = f2bf(h0[r * H_ + wg * 16 + c]);  // phase-0 buffer
    }
    grid_barrier(bar, bar + 1);   // h0 visible everywhere

    const int rt     = wave & 3;          // row tile 0..3 (batch rows rt*16..)
    const int ct0    = (wave >> 2) * 2;   // col tiles {ct0, ct0+1} of 4
    const int arow   = rt * 16 + (lane & 15);
    const int ksel   = (lane < 16) ? 0 : 8;
    const int hi     = (lane < 16) ? 0 : 8;

    for (int t = 0; t < SEQ; ++t) {
        const unsigned short* xrow  = Xb + (size_t)t * BATCH * IN_;
        const unsigned short* hprev = hbuf + (size_t)(t & 1) * (BATCH * H_);
        unsigned short*       hnext = hbuf + (size_t)((t + 1) & 1) * (BATCH * H_);

        v8f acc0 = {0.f,0.f,0.f,0.f,0.f,0.f,0.f,0.f};
        v8f acc1 = {0.f,0.f,0.f,0.f,0.f,0.f,0.f,0.f};

        // ---- K-loop, x part (K = 0..511) ----
        #pragma unroll 4
        for (int k0 = 0; k0 < IN_; k0 += 32) {
            ABfrag a, b0, b1;
            const unsigned short* p = xrow + arow * IN_ + k0 + ksel;
            __builtin_prefetch(p + 128, 0, 1);
            a.u[0] = *(const uint4*)(p);
            a.u[1] = *(const uint4*)(p + 16);
            int ks = k0 >> 5;
            const uint4* f0 = (const uint4*)((const char*)sW + (size_t)(ct0 * 32 + ks) * 1024 + lane * 32);
            const uint4* f1 = (const uint4*)((const char*)sW + (size_t)((ct0 + 1) * 32 + ks) * 1024 + lane * 32);
            b0.u[0] = f0[0]; b0.u[1] = f0[1];
            b1.u[0] = f1[0]; b1.u[1] = f1[1];
            acc0 = __builtin_amdgcn_wmma_f32_16x16x32_bf16(false, a.v, false, b0.v,
                                                           (short)0, acc0, false, false);
            acc1 = __builtin_amdgcn_wmma_f32_16x16x32_bf16(false, a.v, false, b1.v,
                                                           (short)0, acc1, false, false);
        }
        // ---- K-loop, h part (K = 512..1023) ----
        #pragma unroll 4
        for (int k0 = 0; k0 < H_; k0 += 32) {
            ABfrag a, b0, b1;
            const unsigned short* p = hprev + arow * H_ + k0 + ksel;
            a.u[0] = *(const uint4*)(p);
            a.u[1] = *(const uint4*)(p + 16);
            int ks = (k0 + IN_) >> 5;
            const uint4* f0 = (const uint4*)((const char*)sW + (size_t)(ct0 * 32 + ks) * 1024 + lane * 32);
            const uint4* f1 = (const uint4*)((const char*)sW + (size_t)((ct0 + 1) * 32 + ks) * 1024 + lane * 32);
            b0.u[0] = f0[0]; b0.u[1] = f0[1];
            b1.u[0] = f1[0]; b1.u[1] = f1[1];
            acc0 = __builtin_amdgcn_wmma_f32_16x16x32_bf16(false, a.v, false, b0.v,
                                                           (short)0, acc0, false, false);
            acc1 = __builtin_amdgcn_wmma_f32_16x16x32_bf16(false, a.v, false, b1.v,
                                                           (short)0, acc1, false, false);
        }

        // ---- spill gate tiles to LDS: C layout VGPR j -> (M = j + hi, N = lane&15) ----
        {
            int N = lane & 15;
            #pragma unroll
            for (int j = 0; j < 8; ++j) {
                int row = rt * 16 + j + hi;
                sG[row * 64 + ct0 * 16 + N]       = acc0[j];
                sG[row * 64 + (ct0 + 1) * 16 + N] = acc1[j];
            }
        }
        __syncthreads();

        // ---- fused cell update: 64 rows x 16 h-cols ----
        for (int idx = tid; idx < BATCH * 16; idx += THREADS) {
            int r = idx >> 4, c = idx & 15;
            float fg = sigm(sG[r * 64 +  0 + c] + sB[ 0 + c]);
            float ig = sigm(sG[r * 64 + 16 + c] + sB[16 + c]);
            float og = sigm(sG[r * 64 + 32 + c] + sB[32 + c]);
            float gg = tanh_fast(sG[r * 64 + 48 + c] + sB[48 + c]);
            float cv = fg * sC[idx] + ig * gg;
            float hv = og * tanh_fast(cv);
            sC[idx] = cv;
            hnext[r * H_ + wg * 16 + c] = f2bf(hv);
            out[(size_t)t * BATCH * H_ + r * H_ + wg * 16 + c] = hv;
        }

        grid_barrier(bar, bar + 1);   // publish h_t, separate steps
    }

    // ---- tail: h_T and c_T ----
    const size_t outsz = (size_t)SEQ * BATCH * H_;
    for (int idx = tid; idx < BATCH * 16; idx += THREADS) {
        int r = idx >> 4, c = idx & 15;
        size_t off = (size_t)r * H_ + wg * 16 + c;
        out[outsz + off]              = out[(size_t)(SEQ - 1) * BATCH * H_ + off];
        out[outsz + BATCH * H_ + off] = sC[idx];
    }
}

// ---------------- host side ----------------
extern "C" void kernel_launch(void* const* d_in, const int* in_sizes, int n_in,
                              void* d_out, int out_size, void* d_ws, size_t ws_size,
                              hipStream_t stream) {
    const float* x  = (const float*)d_in[0];   // [SEQ,B,IN]
    const float* h0 = (const float*)d_in[1];   // [B,H]
    const float* c0 = (const float*)d_in[2];   // [B,H]
    const float* W  = (const float*)d_in[3];   // [IN+H,4H]
    const float* b  = (const float*)d_in[4];   // [4H]
    float* out = (float*)d_out;

    char* ws = (char*)d_ws;
    unsigned*       bar  = (unsigned*)ws;                          // 2 counters (256 B slot)
    unsigned short* hbuf = (unsigned short*)(ws + 256);            // 2*B*H bf16 = 128 KB
    unsigned short* xb   = (unsigned short*)(ws + 256 + 2 * BATCH * H_ * sizeof(unsigned short));

    const int n8 = SEQ * BATCH * IN_ / 8;                          // 8 floats per thread
    lstm_convert_x<<<n8 / 256, 256, 0, stream>>>(x, xb, bar, n8);

    const size_t smem = 131072 + 16384 + 4096 + 256;               // ~148 KB of 320 KB/WGP
    lstm_seq_wmma<<<NWG, THREADS, smem, stream>>>(W, b, h0, c0, xb, hbuf, out, bar);
}
